// GNN_47098611368430
// MI455X (gfx1250) — matrix operand order repaired and checked by reading, not compiled, not run on hardware
//
#include <hip/hip_runtime.h>

typedef __attribute__((ext_vector_type(16))) _Float16 v16h;
typedef __attribute__((ext_vector_type(8)))  _Float16 v8h;
typedef __attribute__((ext_vector_type(8)))  float    v8f;

// ---------------- small utility kernels ----------------

__global__ void zero_f32(float* __restrict__ p, int n) {
  int i = blockIdx.x * blockDim.x + threadIdx.x;
  if (i < n) p[i] = 0.0f;
}

__global__ void add_vec(const float* __restrict__ a, const float* __restrict__ b,
                        float* __restrict__ o, int n) {
  int i = blockIdx.x * blockDim.x + threadIdx.x;
  if (i < n) o[i] = a[i] + b[i];
}

// f32 (M x D) -> f16 written at columns [col0, col0+D) of an (M x ld) f16 matrix
__global__ void cvt_concat_f16(const float* __restrict__ src, _Float16* __restrict__ dst,
                               int M, int D, int ld, int col0) {
  int i = blockIdx.x * blockDim.x + threadIdx.x;
  if (i >= M * D) return;
  int r = i / D, c = i - r * D;
  dst[(size_t)r * ld + col0 + c] = (_Float16)src[(size_t)r * D + c];
}

// Pack f32 weights (Krows x Nc, row-major, logical K rows starting at krow_off)
// into the WMMA B-fragment layout:
//   dst[((ktile*Nc + n)*2 + half)*16 + e]  holds  W[ktile*32 + half*16 + e][n]
// so a lane (col n, half h) reads its 16 B values as 32 contiguous bytes.
__global__ void pack_w(const float* __restrict__ src, _Float16* __restrict__ dst,
                       int Krows, int Nc, int krow_off) {
  int i = blockIdx.x * blockDim.x + threadIdx.x;
  if (i >= Krows * Nc) return;
  int ks = i / Nc, n = i - ks * Nc;
  int K = krow_off + ks;
  int kt = K >> 5, rem = K & 31;
  int half = rem >> 4, e = rem & 15;
  dst[(((size_t)kt * Nc + n) * 2 + half) * 16 + e] = (_Float16)src[(size_t)ks * Nc + n];
}

// ---------------- SpMM scatter: m[row[e]] += x[col[e]] ----------------
// Edge-parallel, 4 contiguous features per thread; L2-resident float atomics.
__global__ void spmm_scatter(const float* __restrict__ x,
                             const int* __restrict__ erow,
                             const int* __restrict__ ecol,
                             float* __restrict__ m,
                             int E, int D) {
  int t = blockIdx.x * blockDim.x + threadIdx.x;
  int chunks = D >> 2;
  int e = t / chunks;
  int c = (t - e * chunks) << 2;
  if (e >= E) return;
  int src = ecol[e];
  int dst = erow[e];
  const float4 v = *(const float4*)(x + (size_t)src * D + c);
  float* o = m + (size_t)dst * D + c;
  atomicAdd(o + 0, v.x);
  atomicAdd(o + 1, v.y);
  atomicAdd(o + 2, v.z);
  atomicAdd(o + 3, v.w);
}

// ---------------- graph pooling: pooled[idx[n]] += h[n] ----------------
__global__ void pool_scatter(const float* __restrict__ h, const int* __restrict__ idx,
                             float* __restrict__ pooled, int N, int D) {
  int t = blockIdx.x * blockDim.x + threadIdx.x;
  int chunks = D >> 2;
  int n = t / chunks;
  int c = (t - n * chunks) << 2;
  if (n >= N) return;
  int g = idx[n];
  const float4 v = *(const float4*)(h + (size_t)n * D + c);
  float* o = pooled + (size_t)g * D + c;
  atomicAdd(o + 0, v.x);
  atomicAdd(o + 1, v.y);
  atomicAdd(o + 2, v.z);
  atomicAdd(o + 3, v.w);
}

// ---------------- WMMA GEMM ----------------
// C[M x NC] = act( A @ W + bias );  A: (M x KTOT) f16 row-major (rows padded to
// a multiple of 16 so loads need no guard), Wp packed (see pack_w).
// grid.x = M/16 row tiles; blockDim.x = 32*(NC/16): one wave per 16-col tile.
// Optionally also writes an f16 copy of C at column offset coloff16 of an
// (M x ldc16) f16 matrix (feeds the next layer's A without an extra pass).
template <int KTOT, int NC, bool RELU, bool GUARD>
__global__ void gemm_wmma_f16(const _Float16* __restrict__ A,
                              const _Float16* __restrict__ Wp,
                              const float* __restrict__ bias,
                              float* __restrict__ C,
                              _Float16* __restrict__ Cf16, int ldc16, int coloff16,
                              int Mstore) {
  const int wave   = threadIdx.x >> 5;
  const int lane   = threadIdx.x & 31;
  const int halfId = lane >> 4;     // 16-lane half of the wave32
  const int lm     = lane & 15;
  const int row0   = blockIdx.x * 16;
  const int col0   = wave * 16;
  const int mrow   = row0 + lm;     // A row for this lane
  const int ncol   = col0 + lm;     // B/C column for this lane

  const _Float16* __restrict__ arow = A + (size_t)mrow * KTOT;
  v8f acc = {};
#pragma unroll
  for (int kt = 0; kt < KTOT / 32; ++kt) {
    const int k0 = kt * 32;
    // A fragment (16x32 f16): lane holds K = k0 + half*16 + [0..15]
    v16h a = *(const v16h*)(arow + k0 + halfId * 16);
    // B fragment (32x16 f16): 32 contiguous bytes per lane in packed layout
    v16h b = *(const v16h*)(Wp + (((size_t)kt * NC + ncol) * 2 + halfId) * 16);
    acc = __builtin_amdgcn_wmma_f32_16x16x32_f16(
        /*neg_a=*/false, a, /*neg_b=*/false, b,
        /*c_mod=*/(short)0, acc, /*reuse_a=*/false, /*reuse_b=*/false);
  }

  const float bv = bias[ncol];
#pragma unroll
  for (int v = 0; v < 8; ++v) {
    int r = row0 + v + halfId * 8;  // D layout: VGPR v -> row v / v+8 per half
    if (!GUARD || r < Mstore) {
      float val = acc[v] + bv;
      if (RELU) val = fmaxf(val, 0.0f);
      C[(size_t)r * NC + ncol] = val;
      if (Cf16) Cf16[(size_t)r * ldc16 + coloff16 + ncol] = (_Float16)val;
    }
  }
}

// ---------------- host-side orchestration ----------------

static inline int cdiv(int a, int b) { return (a + b - 1) / b; }

extern "C" void kernel_launch(void* const* d_in, const int* in_sizes, int n_in,
                              void* d_out, int out_size, void* d_ws, size_t ws_size,
                              hipStream_t stream) {
  (void)n_in; (void)ws_size;
  const float* x    = (const float*)d_in[0];
  const int*   erow = (const int*)d_in[1];
  const int*   ecol = (const int*)d_in[2];
  const int*   idx  = (const int*)d_in[3];
  const float* w1a  = (const float*)d_in[4];
  const float* b1a  = (const float*)d_in[5];
  const float* w1b  = (const float*)d_in[6];
  const float* b1b  = (const float*)d_in[7];
  const float* w2a  = (const float*)d_in[8];
  const float* b2a  = (const float*)d_in[9];
  const float* w2b  = (const float*)d_in[10];
  const float* b2b  = (const float*)d_in[11];
  const float* wf   = (const float*)d_in[12];
  const float* bf   = (const float*)d_in[13];
  float* out = (float*)d_out;

  const int DIN = 64, DH = 128, DOUT = 64;
  const int N = in_sizes[0] / DIN;      // 50000 (multiple of 16)
  const int E = in_sizes[1];            // 800000
  const int G = out_size / DOUT;        // 500
  const int GPAD = cdiv(G, 16) * 16;    // 512

  // ---- workspace carve-up (f32 first, then f16; all 16B-aligned) ----
  char* ws = (char*)d_ws;
  float* m1     = (float*)ws; ws += (size_t)N * DIN * 4;     // spmm(x)
  float* h1     = (float*)ws; ws += (size_t)N * DH * 4;      // layer1 out (f32, for spmm2)
  float* m2     = (float*)ws; ws += (size_t)N * DH * 4;      // spmm(h1)
  float* h2     = (float*)ws; ws += (size_t)N * DH * 4;      // layer2 out (f32, for pool)
  float* pooled = (float*)ws; ws += (size_t)GPAD * DH * 4;   // padded, zeroed
  float* b1     = (float*)ws; ws += DH * 4;
  float* b2     = (float*)ws; ws += DH * 4;
  _Float16* A1  = (_Float16*)ws; ws += (size_t)N * (2 * DIN) * 2;  // [x | m1] f16
  _Float16* A2  = (_Float16*)ws; ws += (size_t)N * (2 * DH) * 2;   // [h1 | m2] f16
  _Float16* Ap  = (_Float16*)ws; ws += (size_t)GPAD * DH * 2;      // pooled f16
  _Float16* W1  = (_Float16*)ws; ws += (size_t)(2 * DIN) * DH * 2; // packed [w1a;w1b]
  _Float16* W2  = (_Float16*)ws; ws += (size_t)(2 * DH) * DH * 2;  // packed [w2a;w2b]
  _Float16* Wf  = (_Float16*)ws; ws += (size_t)DH * DOUT * 2;      // packed wf

  // ---- weight prep: pack to WMMA B layout, fuse biases ----
  pack_w<<<cdiv(DIN * DH, 256), 256, 0, stream>>>(w1a, W1, DIN, DH, 0);
  pack_w<<<cdiv(DIN * DH, 256), 256, 0, stream>>>(w1b, W1, DIN, DH, DIN);
  pack_w<<<cdiv(DH * DH, 256), 256, 0, stream>>>(w2a, W2, DH, DH, 0);
  pack_w<<<cdiv(DH * DH, 256), 256, 0, stream>>>(w2b, W2, DH, DH, DH);
  pack_w<<<cdiv(DH * DOUT, 256), 256, 0, stream>>>(wf, Wf, DH, DOUT, 0);
  add_vec<<<1, DH, 0, stream>>>(b1a, b1b, b1, DH);
  add_vec<<<1, DH, 0, stream>>>(b2a, b2b, b2, DH);

  // ---- layer 1: h1 = relu([x | spmm(x)] @ W1 + b1) ----
  zero_f32<<<cdiv(N * DIN, 256), 256, 0, stream>>>(m1, N * DIN);
  spmm_scatter<<<cdiv(E * (DIN / 4), 256), 256, 0, stream>>>(x, erow, ecol, m1, E, DIN);
  cvt_concat_f16<<<cdiv(N * DIN, 256), 256, 0, stream>>>(x, A1, N, DIN, 2 * DIN, 0);
  cvt_concat_f16<<<cdiv(N * DIN, 256), 256, 0, stream>>>(m1, A1, N, DIN, 2 * DIN, DIN);
  gemm_wmma_f16<128, 128, true, false><<<N / 16, 256, 0, stream>>>(
      A1, W1, b1, h1, A2, 2 * DH, 0, N);   // f16 copy of h1 -> A2 cols [0,128)

  // ---- layer 2: h2 = relu([h1 | spmm(h1)] @ W2 + b2) ----
  zero_f32<<<cdiv(N * DH, 256), 256, 0, stream>>>(m2, N * DH);
  spmm_scatter<<<cdiv(E * (DH / 4), 256), 256, 0, stream>>>(h1, erow, ecol, m2, E, DH);
  cvt_concat_f16<<<cdiv(N * DH, 256), 256, 0, stream>>>(m2, A2, N, DH, 2 * DH, DH);
  gemm_wmma_f16<256, 128, true, false><<<N / 16, 256, 0, stream>>>(
      A2, W2, b2, h2, (_Float16*)nullptr, 0, 0, N);

  // ---- readout: out = segment_sum(h2, idx) @ wf + bf ----
  zero_f32<<<cdiv(GPAD * DH, 256), 256, 0, stream>>>(pooled, GPAD * DH);
  pool_scatter<<<cdiv(N * (DH / 4), 256), 256, 0, stream>>>(h2, idx, pooled, N, DH);
  cvt_concat_f16<<<cdiv(GPAD * DH, 256), 256, 0, stream>>>(pooled, Ap, GPAD, DH, DH, 0);
  gemm_wmma_f16<128, 64, false, true><<<GPAD / 16, 128, 0, stream>>>(
      Ap, Wf, bf, out, (_Float16*)nullptr, 0, 0, G);
}
